// Graph_Model_39685497815515
// MI455X (gfx1250) — matrix operand order repaired and checked by reading, not compiled, not run on hardware
//
#include <hip/hip_runtime.h>

// ---------------- problem constants ----------------
#define N_NODES 10000
#define N_EDGES 160000
#define NREL    4
#define DIN     15
#define HID     512
#define NGRAPH  64
#define MT      625      // 10000 / 16 exactly
#define KCAT1   96       // [x(16) | mean_r*16 (64)] padded to 96
#define KT1     3        // 96/32
#define KCAT2   2560     // [h(512) | mean_r*512 (2048)]
#define KT2     80       // 2560/32

typedef __attribute__((ext_vector_type(16))) __bf16 v16bf;
typedef __attribute__((ext_vector_type(8)))  float  v8f;

// float -> bf16 bits, round-to-nearest-even
__device__ __forceinline__ unsigned short f2bf(float f) {
    unsigned int u = __float_as_uint(f);
    u += 0x7FFFu + ((u >> 16) & 1u);
    return (unsigned short)(u >> 16);
}

__device__ __forceinline__ unsigned int pack2bf(float lo, float hi) {
    return (unsigned int)f2bf(lo) | ((unsigned int)f2bf(hi) << 16);
}

// WMMA 16-bit A-matrix (16x32) lane record (ISA 7.12.2):
// record = ((mt*KT + kt)*32 + lane) * 16 u16 elements; lane = 16*half + mrow.
// element i in 0..7  -> K = kt*32 +      half*8 + i   (row M = mrow)
// element i in 8..15 -> K = kt*32 + 16 + half*8 + (i-8)

// ---------------- kernels ----------------
__global__ void k_zero(float* p, long n) {
    long i = (long)blockIdx.x * blockDim.x + threadIdx.x;
    long stride = (long)gridDim.x * blockDim.x;
    for (; i < n; i += stride) p[i] = 0.0f;
}

// layer-1 edge scatter: seg = tgt*NREL + type; cnt += 1; sums1[seg][0..14] += x[src]
__global__ void k_scatter1(const int* __restrict__ src, const int* __restrict__ tgt,
                           const int* __restrict__ et, const float* __restrict__ x,
                           float* __restrict__ cnt, float* __restrict__ sums1) {
    int e = blockIdx.x * blockDim.x + threadIdx.x;
    if (e >= N_EDGES) return;
    int s = src[e];
    int seg = tgt[e] * NREL + et[e];
    atomicAdd(&cnt[seg], 1.0f);
    const float* xs = x + s * DIN;
    float* o = sums1 + seg * 16;
#pragma unroll
    for (int d = 0; d < DIN; ++d) atomicAdd(&o[d], xs[d]);
}

__global__ void k_gcnt(const int* __restrict__ batch, float* __restrict__ cntG) {
    int n = blockIdx.x * blockDim.x + threadIdx.x;
    if (n >= N_NODES) return;
    atomicAdd(&cntG[batch[n]], 1.0f);
}

// pack layer-1 features into WMMA A layout, one 32B lane record per thread
__global__ void k_packA1(const float* __restrict__ x, const float* __restrict__ sums1,
                         const float* __restrict__ cnt, unsigned short* __restrict__ A) {
    int rec = blockIdx.x * blockDim.x + threadIdx.x;
    if (rec >= MT * KT1 * 32) return;
    int lane = rec & 31;
    int t    = rec >> 5;
    int kt   = t % KT1;
    int mt   = t / KT1;
    int half = lane >> 4, mrow = lane & 15;
    int n = mt * 16 + mrow;
    unsigned int w[8];
#pragma unroll
    for (int g = 0; g < 2; ++g) {
        int k0  = kt * 32 + g * 16 + half * 8;   // 8 contiguous K, inside one 16-block
        int b16 = k0 >> 4;                       // 0:x, 1..4:mean_r, 5:pad
        int off = k0 & 15;                       // 0 or 8
        float f[8];
        if (b16 == 0) {
#pragma unroll
            for (int j = 0; j < 8; ++j) {
                int d = off + j;
                f[j] = (d < DIN) ? x[n * DIN + d] : 0.0f;
            }
        } else if (b16 <= 4) {
            int seg = n * NREL + (b16 - 1);
            float inv = 1.0f / fmaxf(cnt[seg], 1.0f);
            const float* sp = sums1 + seg * 16 + off;   // pad slot 15 stays 0
#pragma unroll
            for (int j = 0; j < 8; ++j) f[j] = sp[j] * inv;
        } else {
#pragma unroll
            for (int j = 0; j < 8; ++j) f[j] = 0.0f;
        }
#pragma unroll
        for (int j = 0; j < 4; ++j) w[g * 4 + j] = pack2bf(f[2 * j], f[2 * j + 1]);
    }
    uint4* outp = (uint4*)(A + rec * 16);
    outp[0] = make_uint4(w[0], w[1], w[2], w[3]);
    outp[1] = make_uint4(w[4], w[5], w[6], w[7]);
}

// pack Wcat1 = [W1_root(15)+pad | W1_rel r0..r3 (15+pad each) | pad16] into WMMA B layout
__global__ void k_packB1(const float* __restrict__ W1_rel, const float* __restrict__ W1_root,
                         unsigned short* __restrict__ B) {
    int id = blockIdx.x * blockDim.x + threadIdx.x;
    if (id >= 32 * KT1 * 32 * 16) return;
    int i = id & 15;
    int rem = id >> 4;
    int lane = rem & 31;
    int rem2 = rem >> 5;
    int kt = rem2 % KT1;
    int nt = rem2 / KT1;
    int K = kt * 32 + ((i >> 3) * 16) + ((lane >> 4) * 8) + (i & 7);
    int col = nt * 16 + (lane & 15);
    float v = 0.0f;
    if (K < DIN) v = W1_root[K * HID + col];
    else if (K >= 16 && K < 80) {
        int r = (K - 16) >> 4, d = (K - 16) & 15;
        if (d < DIN) v = W1_rel[(r * DIN + d) * HID + col];
    }
    B[id] = f2bf(v);
}

// GEMM1: h1[10000,512] = relu(A @ B + b1); LDS-staged A tile, 8 waves x 4 N-tiles
__global__ __launch_bounds__(256) void k_gemm1(const unsigned short* __restrict__ A,
                                               const unsigned short* __restrict__ B,
                                               const float* __restrict__ bias,
                                               float* __restrict__ out) {
    __shared__ unsigned short As[2][512];      // 1KB tile, double-buffered
    int mt = blockIdx.x;
    int wave = threadIdx.x >> 5;
    int lane = threadIdx.x & 31;
    const unsigned int* A32 = (const unsigned int*)A;
    v8f acc[4] = {};
    for (int kt = 0; kt < KT1; ++kt) {
        ((unsigned int*)As[kt & 1])[threadIdx.x] = A32[(mt * KT1 + kt) * 256 + threadIdx.x];
        __syncthreads();
        v16bf a = *(const v16bf*)(&As[kt & 1][lane * 16]);
#pragma unroll
        for (int t = 0; t < 4; ++t) {
            int nt = wave + t * 8;
            v16bf b = *(const v16bf*)(B + ((nt * KT1 + kt) * 32 + lane) * 16);
            acc[t] = __builtin_amdgcn_wmma_f32_16x16x32_bf16(false, a, false, b,
                                                             (short)0, acc[t], false, false);
        }
    }
    int half = lane >> 4, nn = lane & 15;
#pragma unroll
    for (int t = 0; t < 4; ++t) {
        int col = (wave + t * 8) * 16 + nn;
        float bv = bias[col];
#pragma unroll
        for (int v = 0; v < 8; ++v) {
            int row = mt * 16 + v + half * 8;
            out[row * HID + col] = fmaxf(acc[t][v] + bv, 0.0f);
        }
    }
}

// layer-2 edge scatter: sums2[seg][0..511] += h1[src]; thread = (edge, 4-float chunk)
__global__ void k_scatter2(const int* __restrict__ src, const int* __restrict__ tgt,
                           const int* __restrict__ et, const float* __restrict__ h1,
                           float* __restrict__ sums2) {
    int id = blockIdx.x * blockDim.x + threadIdx.x;
    if (id >= N_EDGES * 128) return;
    int e = id >> 7;
    int c = (id & 127) << 2;
    int s = src[e];
    int seg = tgt[e] * NREL + et[e];
    const float4 v = *(const float4*)(h1 + s * HID + c);
    float* o = sums2 + (long)seg * HID + c;
    atomicAdd(o + 0, v.x);
    atomicAdd(o + 1, v.y);
    atomicAdd(o + 2, v.z);
    atomicAdd(o + 3, v.w);
}

// pack layer-2 features [h | mean_r0..r3] into WMMA A layout, 32B record per thread
__global__ void k_packA2(const float* __restrict__ h1, const float* __restrict__ sums2,
                         const float* __restrict__ cnt, unsigned short* __restrict__ A) {
    int rec = blockIdx.x * blockDim.x + threadIdx.x;
    if (rec >= MT * KT2 * 32) return;
    int lane = rec & 31;
    int t    = rec >> 5;
    int kt   = t % KT2;
    int mt   = t / KT2;
    int half = lane >> 4, mrow = lane & 15;
    int n = mt * 16 + mrow;
    unsigned int w[8];
#pragma unroll
    for (int g = 0; g < 2; ++g) {
        int k0  = kt * 32 + g * 16 + half * 8;  // 8 contiguous K, inside one 512-block
        int blk = k0 >> 9, d0 = k0 & 511;
        const float* sp;
        float scale = 1.0f;
        if (blk == 0) {
            sp = h1 + n * HID + d0;
        } else {
            int seg = n * NREL + (blk - 1);
            sp = sums2 + (long)seg * HID + d0;
            scale = 1.0f / fmaxf(cnt[seg], 1.0f);
        }
        float4 v0 = *(const float4*)(sp);
        float4 v1 = *(const float4*)(sp + 4);
        w[g * 4 + 0] = pack2bf(v0.x * scale, v0.y * scale);
        w[g * 4 + 1] = pack2bf(v0.z * scale, v0.w * scale);
        w[g * 4 + 2] = pack2bf(v1.x * scale, v1.y * scale);
        w[g * 4 + 3] = pack2bf(v1.z * scale, v1.w * scale);
    }
    uint4* outp = (uint4*)(A + rec * 16);
    outp[0] = make_uint4(w[0], w[1], w[2], w[3]);
    outp[1] = make_uint4(w[4], w[5], w[6], w[7]);
}

// pack Wcat2 = [W2_root ; W2_rel r0..r3] into WMMA B layout
__global__ void k_packB2(const float* __restrict__ W2_rel, const float* __restrict__ W2_root,
                         unsigned short* __restrict__ B) {
    int id = blockIdx.x * blockDim.x + threadIdx.x;
    if (id >= 32 * KT2 * 32 * 16) return;
    int i = id & 15;
    int rem = id >> 4;
    int lane = rem & 31;
    int rem2 = rem >> 5;
    int kt = rem2 % KT2;
    int nt = rem2 / KT2;
    int K = kt * 32 + ((i >> 3) * 16) + ((lane >> 4) * 8) + (i & 7);
    int col = nt * 16 + (lane & 15);
    int blk = K >> 9, d = K & 511;
    float v = (blk == 0) ? W2_root[d * HID + col]
                         : W2_rel[((long)(blk - 1) * HID + d) * HID + col];
    B[id] = f2bf(v);
}

// GEMM2 with LDS-staged A and fused bias+ReLU+graph-pool epilogue
__global__ __launch_bounds__(256) void k_gemm2(const unsigned short* __restrict__ A,
                                               const unsigned short* __restrict__ B,
                                               const float* __restrict__ bias,
                                               const int* __restrict__ batch,
                                               float* __restrict__ pool) {
    __shared__ unsigned short As[2][512];
    int mt = blockIdx.x;
    int wave = threadIdx.x >> 5;
    int lane = threadIdx.x & 31;
    const unsigned int* A32 = (const unsigned int*)A;
    v8f acc[4] = {};
    for (int kt = 0; kt < KT2; ++kt) {
        ((unsigned int*)As[kt & 1])[threadIdx.x] = A32[(mt * KT2 + kt) * 256 + threadIdx.x];
        __syncthreads();
        v16bf a = *(const v16bf*)(&As[kt & 1][lane * 16]);
#pragma unroll
        for (int t = 0; t < 4; ++t) {
            int nt = wave + t * 8;
            v16bf b = *(const v16bf*)(B + ((nt * KT2 + kt) * 32 + lane) * 16);
            acc[t] = __builtin_amdgcn_wmma_f32_16x16x32_bf16(false, a, false, b,
                                                             (short)0, acc[t], false, false);
        }
    }
    int half = lane >> 4, nn = lane & 15;
#pragma unroll
    for (int v = 0; v < 8; ++v) {
        int row = mt * 16 + v + half * 8;
        int g = batch[row];
#pragma unroll
        for (int t = 0; t < 4; ++t) {
            int col = (wave + t * 8) * 16 + nn;
            float val = fmaxf(acc[t][v] + bias[col], 0.0f);
            atomicAdd(&pool[g * HID + col], val);
        }
    }
}

__global__ void k_final(const float* __restrict__ pool, const float* __restrict__ cntG,
                        float* __restrict__ out) {
    int id = blockIdx.x * blockDim.x + threadIdx.x;
    if (id >= NGRAPH * HID) return;
    int g = id / HID;
    out[id] = pool[id] / fmaxf(cntG[g], 1.0f);
}

// ---------------- host launch ----------------
extern "C" void kernel_launch(void* const* d_in, const int* in_sizes, int n_in,
                              void* d_out, int out_size, void* d_ws, size_t ws_size,
                              hipStream_t stream) {
    const float* x       = (const float*)d_in[0];
    const int*   eidx    = (const int*)  d_in[1];
    const int*   etype   = (const int*)  d_in[2];
    const int*   batch   = (const int*)  d_in[3];
    const float* W1_rel  = (const float*)d_in[4];
    const float* W1_root = (const float*)d_in[5];
    const float* b1      = (const float*)d_in[6];
    const float* W2_rel  = (const float*)d_in[7];
    const float* W2_root = (const float*)d_in[8];
    const float* b2      = (const float*)d_in[9];
    float* out = (float*)d_out;

    const int* src = eidx;
    const int* tgt = eidx + N_EDGES;

    // ---- workspace carving (float region first, then bf16/u16 region) ----
    float* wsf = (float*)d_ws;
    float* cnt   = wsf;                         // N*R            = 40000
    float* cntG  = cnt   + 40000;               // 64
    float* pool  = cntG  + 64;                  // 64*512         = 32768
    float* sums1 = pool  + 32768;               // N*R*16         = 640000
    float* sums2 = sums1 + 640000;              // N*R*512        = 20480000
    const long n_zero = 40000L + 64 + 32768 + 640000 + 20480000;   // 21,192,832 floats
    float* h1 = wsf + n_zero;                   // N*512          = 5,120,000
    unsigned short* wsu = (unsigned short*)(h1 + (long)N_NODES * HID);
    unsigned short* featA1 = wsu;                         // 625*3*512    =   960,000
    unsigned short* Bp1    = featA1 + 960000L;            // 32*3*32*16   =    49,152
    unsigned short* featA2 = Bp1    + 49152L;             // 625*80*512   = 25,600,000
    unsigned short* Bp2    = featA2 + 25600000L;          // 32*80*32*16  =  1,310,720

    // ---- pipeline (stream-ordered) ----
    k_zero<<<8192, 256, 0, stream>>>(wsf, n_zero);

    k_scatter1<<<(N_EDGES + 255) / 256, 256, 0, stream>>>(src, tgt, etype, x, cnt, sums1);
    k_gcnt<<<(N_NODES + 255) / 256, 256, 0, stream>>>(batch, cntG);
    k_packB1<<<(32 * KT1 * 32 * 16 + 255) / 256, 256, 0, stream>>>(W1_rel, W1_root, Bp1);
    k_packB2<<<(32 * KT2 * 32 * 16 + 255) / 256, 256, 0, stream>>>(W2_rel, W2_root, Bp2);

    k_packA1<<<(MT * KT1 * 32 + 255) / 256, 256, 0, stream>>>(x, sums1, cnt, featA1);
    k_gemm1<<<MT, 256, 0, stream>>>(featA1, Bp1, b1, h1);

    k_scatter2<<<(N_EDGES * 128 + 255) / 256, 256, 0, stream>>>(src, tgt, etype, h1, sums2);
    k_packA2<<<(MT * KT2 * 32 + 255) / 256, 256, 0, stream>>>(h1, sums2, cnt, featA2);
    k_gemm2<<<MT, 256, 0, stream>>>(featA2, Bp2, b2, batch, pool);

    k_final<<<(NGRAPH * HID + 255) / 256, 256, 0, stream>>>(pool, cntG, out);
}
